// Model_39676907881574
// MI455X (gfx1250) — matrix operand order repaired and checked by reading, not compiled, not run on hardware
//
#include <hip/hip_runtime.h>

typedef __attribute__((ext_vector_type(16))) _Float16 v16h;
typedef __attribute__((ext_vector_type(8)))  _Float16 v8h;
typedef __attribute__((ext_vector_type(4)))  _Float16 v4h;
typedef __attribute__((ext_vector_type(8)))  float    v8f;

#define B_      2
#define H_      16
#define S_      2048
#define D_      128
#define BLOCK_N 32     // keys per iteration
#define BLOCK_M 64     // query rows per workgroup (16 per wave)
#define NWAVES  4
#define SM_SCALE 0.08838834764831845f  // 1/sqrt(128)
#define STG     8      // float4 staging elements per thread: (32*128/4)/128

__launch_bounds__(128, 1)
__global__ void fa_fwd_f16wmma(const float* __restrict__ Q,
                               const float* __restrict__ K,
                               const float* __restrict__ V,
                               float* __restrict__ O) {
    // Double-buffered K (row-major f16 [key][d]) and V (transposed f16 [d][key]),
    // plus wave-private P tiles. 2*8KB + 2*8KB + 4KB = 36KB of 320KB.
    __shared__ _Float16 lds_k[2][BLOCK_N * D_];
    __shared__ _Float16 lds_v[2][D_ * BLOCK_N];
    __shared__ _Float16 lds_p[NWAVES * 16 * BLOCK_N];

    const int tid  = threadIdx.x;
    const int lane = tid & 31;
    const int wv   = tid >> 5;
    const int bh   = blockIdx.x / (S_ / BLOCK_M);   // 0..31 = b*H + h
    const int qt   = blockIdx.x % (S_ / BLOCK_M);

    const size_t base = (size_t)bh * S_ * D_;
    const float* Kb = K + base;
    const float* Vb = V + base;

    const int q0 = qt * BLOCK_M + wv * 16;
    const int lh = lane & 15;                 // column / row-in-tile index
    const int c0 = (lane >> 4) * 8;           // A-layout K-chunk base (0 or 8)
    const int cb = (lane >> 4) * 16;          // contiguous-order K base (0 or 16); unused, B uses c0
    const int rh = (lane >> 4) * 8;           // C-layout row-half offset (0 or 8)

    // staging coordinates for this thread (same for every block)
    int skey[STG], sd[STG];
    #pragma unroll
    for (int i = 0; i < STG; ++i) {
        const int f = tid + i * 128;
        skey[i] = f >> 5;          // 0..31
        sd[i]   = (f & 31) * 4;    // 0..124, step 4
    }

    // ---- preload Q tile in A-layout fragments (4 slices of K=32) ----
    v16h aq[4];
    {
        const float* qrow = Q + base + (size_t)(q0 + lh) * D_;
        #pragma unroll
        for (int s = 0; s < 4; ++s) {
            const int d0 = s * 32 + c0;
            #pragma unroll
            for (int j = 0; j < 8; ++j) {
                aq[s][j]     = (_Float16)qrow[d0 + j];
                aq[s][j + 8] = (_Float16)qrow[d0 + 16 + j];
            }
        }
    }

    // ---- prologue: stage key-block 0 into buffer 0 ----
    #pragma unroll
    for (int i = 0; i < STG; ++i) {
        const size_t g = (size_t)skey[i] * D_ + sd[i];
        const float4 kk = *reinterpret_cast<const float4*>(Kb + g);
        v4h kh;
        kh[0] = (_Float16)kk.x; kh[1] = (_Float16)kk.y;
        kh[2] = (_Float16)kk.z; kh[3] = (_Float16)kk.w;
        *reinterpret_cast<v4h*>(&lds_k[0][skey[i] * D_ + sd[i]]) = kh;

        const float4 vvv = *reinterpret_cast<const float4*>(Vb + g);
        lds_v[0][(sd[i] + 0) * BLOCK_N + skey[i]] = (_Float16)vvv.x;
        lds_v[0][(sd[i] + 1) * BLOCK_N + skey[i]] = (_Float16)vvv.y;
        lds_v[0][(sd[i] + 2) * BLOCK_N + skey[i]] = (_Float16)vvv.z;
        lds_v[0][(sd[i] + 3) * BLOCK_N + skey[i]] = (_Float16)vvv.w;
    }

    v8f acc[8];
    #pragma unroll
    for (int dt = 0; dt < 8; ++dt)
        #pragma unroll
        for (int r = 0; r < 8; ++r) acc[dt][r] = 0.0f;

    float m_r[8], l_r[8];
    #pragma unroll
    for (int r = 0; r < 8; ++r) { m_r[r] = -1e30f; l_r[r] = 0.0f; }

    const int pbase = wv * 16 * BLOCK_N;
    int p = 0;

    for (int n0 = 0; n0 < S_; n0 += BLOCK_N, p ^= 1) {
        __syncthreads();  // staged buf[p] visible; prior reads of buf[p^1] done

        // ---- issue next block's global loads; held in regs across GEMM1 ----
        const bool has_next = (n0 + BLOCK_N) < S_;
        float4 kreg[STG], vreg[STG];
        if (has_next) {
            #pragma unroll
            for (int i = 0; i < STG; ++i) {
                const size_t g = (size_t)(n0 + BLOCK_N + skey[i]) * D_ + sd[i];
                kreg[i] = *reinterpret_cast<const float4*>(Kb + g);
                vreg[i] = *reinterpret_cast<const float4*>(Vb + g);
            }
        }

        // ---- GEMM1: scores = Q @ K^T for two 16-key halves ----
        v8f sc0 = {}, sc1 = {};
        #pragma unroll
        for (int sl = 0; sl < 4; ++sl) {
            {
                const int off = lh * D_ + sl * 32 + c0;
                v8h lo = *reinterpret_cast<const v8h*>(&lds_k[p][off]);
                v8h hi = *reinterpret_cast<const v8h*>(&lds_k[p][off + 16]);
                v16h b = __builtin_shufflevector(lo, hi, 0,1,2,3,4,5,6,7,8,9,10,11,12,13,14,15);
                sc0 = __builtin_amdgcn_wmma_f32_16x16x32_f16(false, aq[sl], false, b,
                                                             (short)0, sc0, false, false);
            }
            {
                const int off = (16 + lh) * D_ + sl * 32 + c0;
                v8h lo = *reinterpret_cast<const v8h*>(&lds_k[p][off]);
                v8h hi = *reinterpret_cast<const v8h*>(&lds_k[p][off + 16]);
                v16h b = __builtin_shufflevector(lo, hi, 0,1,2,3,4,5,6,7,8,9,10,11,12,13,14,15);
                sc1 = __builtin_amdgcn_wmma_f32_16x16x32_f16(false, aq[sl], false, b,
                                                             (short)0, sc1, false, false);
            }
        }

        // ---- online softmax (per C-layout row r + rh) ----
        float alpha[8];
        #pragma unroll
        for (int r = 0; r < 8; ++r) {
            float x0 = sc0[r] * SM_SCALE;
            float x1 = sc1[r] * SM_SCALE;
            float v = fmaxf(x0, x1);
            v = fmaxf(v, __shfl_xor(v, 1, 32));
            v = fmaxf(v, __shfl_xor(v, 2, 32));
            v = fmaxf(v, __shfl_xor(v, 4, 32));
            v = fmaxf(v, __shfl_xor(v, 8, 32));
            const float mn = fmaxf(m_r[r], v);
            const float a  = __expf(m_r[r] - mn);
            const float p0 = __expf(x0 - mn);
            const float p1 = __expf(x1 - mn);
            float t = p0 + p1;
            t += __shfl_xor(t, 1, 32);
            t += __shfl_xor(t, 2, 32);
            t += __shfl_xor(t, 4, 32);
            t += __shfl_xor(t, 8, 32);
            l_r[r] = l_r[r] * a + t;
            m_r[r] = mn;
            alpha[r] = a;
            const int row = r + rh;
            lds_p[pbase + row * BLOCK_N + lh]      = (_Float16)p0;
            lds_p[pbase + row * BLOCK_N + 16 + lh] = (_Float16)p1;
        }

        // rescale accumulators by exp(m_old - m_new)
        #pragma unroll
        for (int dt = 0; dt < 8; ++dt)
            #pragma unroll
            for (int r = 0; r < 8; ++r) acc[dt][r] *= alpha[r];

        __syncthreads();  // P visible; all GEMM1 reads of buf[p] done

        // ---- stage next block into buf[p^1]; ds-stores overlap GEMM2 wmma ----
        if (has_next) {
            #pragma unroll
            for (int i = 0; i < STG; ++i) {
                v4h kh;
                kh[0] = (_Float16)kreg[i].x; kh[1] = (_Float16)kreg[i].y;
                kh[2] = (_Float16)kreg[i].z; kh[3] = (_Float16)kreg[i].w;
                *reinterpret_cast<v4h*>(&lds_k[p ^ 1][skey[i] * D_ + sd[i]]) = kh;

                lds_v[p ^ 1][(sd[i] + 0) * BLOCK_N + skey[i]] = (_Float16)vreg[i].x;
                lds_v[p ^ 1][(sd[i] + 1) * BLOCK_N + skey[i]] = (_Float16)vreg[i].y;
                lds_v[p ^ 1][(sd[i] + 2) * BLOCK_N + skey[i]] = (_Float16)vreg[i].z;
                lds_v[p ^ 1][(sd[i] + 3) * BLOCK_N + skey[i]] = (_Float16)vreg[i].w;
            }
        }

        // ---- read P back in A layout (16 x 32 keys) ----
        v16h ap;
        {
            const int aoff = pbase + lh * BLOCK_N + c0;
            v8h lo = *reinterpret_cast<const v8h*>(&lds_p[aoff]);
            v8h hi = *reinterpret_cast<const v8h*>(&lds_p[aoff + 16]);
            ap = __builtin_shufflevector(lo, hi, 0,1,2,3,4,5,6,7,8,9,10,11,12,13,14,15);
        }

        // ---- GEMM2: acc += P @ V  (8 d-tiles of 16) ----
        #pragma unroll
        for (int dt = 0; dt < 8; ++dt) {
            const int off = (dt * 16 + lh) * BLOCK_N + c0;
            v8h lo = *reinterpret_cast<const v8h*>(&lds_v[p][off]);
            v8h hi = *reinterpret_cast<const v8h*>(&lds_v[p][off + 16]);
            v16h b = __builtin_shufflevector(lo, hi, 0,1,2,3,4,5,6,7,8,9,10,11,12,13,14,15);
            acc[dt] = __builtin_amdgcn_wmma_f32_16x16x32_f16(false, ap, false, b,
                                                             (short)0, acc[dt], false, false);
        }
    }

    // ---- finalize: out = acc / l, fp32 store ----
    #pragma unroll
    for (int r = 0; r < 8; ++r) {
        const float inv_l = 1.0f / l_r[r];
        const size_t row_off = base + (size_t)(q0 + r + rh) * D_;
        #pragma unroll
        for (int dt = 0; dt < 8; ++dt) {
            O[row_off + dt * 16 + lh] = acc[dt][r] * inv_l;
        }
    }
}

extern "C" void kernel_launch(void* const* d_in, const int* in_sizes, int n_in,
                              void* d_out, int out_size, void* d_ws, size_t ws_size,
                              hipStream_t stream) {
    const float* Q = (const float*)d_in[0];
    const float* K = (const float*)d_in[1];
    const float* V = (const float*)d_in[2];
    float* O = (float*)d_out;

    dim3 grid(B_ * H_ * (S_ / BLOCK_M));  // 1024 workgroups
    dim3 block(128);                      // 4 waves (wave32)
    fa_fwd_f16wmma<<<grid, block, 0, stream>>>(Q, K, V, O);
}
